// InterpretableMultiHeadSelfAttention_46755013984772
// MI455X (gfx1250) — matrix operand order repaired and checked by reading, not compiled, not run on hardware
//
#include <hip/hip_runtime.h>
#include <hip/hip_bf16.h>
#include <cstdint>

#define HEADS 8
#define BSZ   4
#define TSEQ  2048
#define UNITS 1024
#define DATT  128

typedef __attribute__((ext_vector_type(16))) __bf16 v16bf;
typedef __attribute__((ext_vector_type(8)))  __bf16 v8bf;
typedef __attribute__((ext_vector_type(8)))  float  v8f;

// ---------------------------------------------------------------------------
// Fragment loaders following CDNA5 ISA 7.12.2 VGPR layouts (wave32).
// A (16x32 bf16, MxK): lane l<16 -> row l, K {0..7,16..23}; lane l>=16 -> row
// l-16, K {8..15,24..31}.
// ---------------------------------------------------------------------------
static __device__ __forceinline__ v16bf load_a_frag(const __bf16* base, int stride) {
  const int l   = threadIdx.x & 31;
  const int row = l & 15;
  const int kb  = (l >> 4) << 3;
  const __bf16* p = base + (size_t)row * stride + kb;
  v8bf x0 = *(const v8bf*)(p);
  v8bf x1 = *(const v8bf*)(p + 16);
  v16bf r;
#pragma unroll
  for (int i = 0; i < 8; ++i) { r[i] = x0[i]; r[i + 8] = x1[i]; }
  return r;
}

// B (32x16 bf16, KxN): lane l<16 -> column l, K {0..7,16..23}; lane l>=16 -> column l-16, K {8..15,24..31}.
static __device__ __forceinline__ v16bf load_b_frag(const __bf16* base, int stride) {
  const int l   = threadIdx.x & 31;
  const int col = l & 15;
  const int kr  = (l >> 4) << 3;
  const __bf16* p = base + (size_t)col * stride + kr;
  v8bf x0 = *(const v8bf*)(p);
  v8bf x1 = *(const v8bf*)(p + 16);
  v16bf r;
#pragma unroll
  for (int i = 0; i < 8; ++i) { r[i] = x0[i]; r[i + 8] = x1[i]; }
  return r;
}

static __device__ __forceinline__ v8f wmma_bf16(v16bf a, v16bf b, v8f c) {
  return __builtin_amdgcn_wmma_f32_16x16x32_bf16(false, a, false, b, (short)0, c,
                                                 false, false);
}

// CDNA5 async global->LDS copy (VGLOBAL opcode 98, GV mode), ASYNCcnt-tracked.
// lds_dst generic pointer: low 32 bits are the LDS byte address (ISA 10.2).
static __device__ __forceinline__ void async_copy_b128(void* lds_dst,
                                                       const void* gsrc) {
  unsigned loff = (unsigned)(uintptr_t)lds_dst;
  asm volatile("global_load_async_to_lds_b128 %0, %1, off"
               :: "v"(loff), "v"(gsrc)
               : "memory");
}
static __device__ __forceinline__ void async_wait0() {
  asm volatile("s_wait_asynccnt 0x0" ::: "memory");
}

// ---------------------------------------------------------------------------
// Elementwise cast kernels
// ---------------------------------------------------------------------------
__global__ void cast_f32_bf16_kernel(const float* __restrict__ in,
                                     __bf16* __restrict__ out, int n) {
  int i = blockIdx.x * blockDim.x + threadIdx.x;
  if (i < n) out[i] = (__bf16)in[i];
}

// in: [K][N] f32 row-major -> out: [N][K] bf16 (B fragments contiguous-in-K)
__global__ void transpose_cast_kernel(const float* __restrict__ in,
                                      __bf16* __restrict__ out, int K, int N,
                                      float scale) {
  int i = blockIdx.x * blockDim.x + threadIdx.x;
  if (i >= K * N) return;
  int k = i / N, n = i % N;
  out[(size_t)n * K + k] = (__bf16)(in[i] * scale);
}

// ---------------------------------------------------------------------------
// WMMA GEMM: C[M x N] = A[M x K] * Bt[N x K]^T, A/Bt bf16, acc f32.
// Block = 256 threads (8 waves); block tile 256(M) x 64(N); wave tile 32x64.
// The 64x32 B tile (shared by all 8 waves) is staged through LDS with
// double-buffered async global->LDS b128 copies (one per thread per K step),
// so the prefetch of step k+1 overlaps the 8 WMMAs of step k.
// mode 0: f32 row-major [M][N] out
// mode 1: bf16 scattered to Q/K layout [B,H,T,DATT]
// mode 2: bf16 scattered to Vt layout [B,DATT,T] (N==128)
// ---------------------------------------------------------------------------
__global__ void gemm_bf16_kernel(const __bf16* __restrict__ A,
                                 const __bf16* __restrict__ Bt,
                                 int N, int K, void* __restrict__ outp, int mode) {
  __shared__ alignas(16) __bf16 Bs[2][64 * 32];

  const int tid  = threadIdx.x;
  const int wave = tid >> 5;
  const int m0 = blockIdx.x * 256 + wave * 32;
  const int n0 = blockIdx.y * 64;

  // Per-thread staging slice: 16 bytes.  LDS tile layout [n-row][k] row-major;
  // element offset srow*32 + skoff == tid*8, i.e. linear in tid.
  const int srow  = tid >> 2;        // 0..63
  const int skoff = (tid & 3) << 3;  // 0,8,16,24

  // stage k-step 0 into buffer 0
  async_copy_b128(&Bs[0][tid * 8], Bt + (size_t)(n0 + srow) * K + skoff);

  v8f c[2][4];
#pragma unroll
  for (int ms = 0; ms < 2; ++ms)
#pragma unroll
    for (int j = 0; j < 4; ++j) c[ms][j] = v8f{};

  int buf = 0;
  for (int k0 = 0; k0 < K; k0 += 32) {
    async_wait0();
    __syncthreads();  // staged tile visible to all waves
    if (k0 + 32 < K)  // prefetch next K step into the other buffer
      async_copy_b128(&Bs[buf ^ 1][tid * 8],
                      Bt + (size_t)(n0 + srow) * K + (k0 + 32) + skoff);

    v16bf a0 = load_a_frag(A + (size_t)m0 * K + k0, K);
    v16bf a1 = load_a_frag(A + (size_t)(m0 + 16) * K + k0, K);
#pragma unroll
    for (int j = 0; j < 4; ++j) {
      v16bf b = load_b_frag(&Bs[buf][j * 16 * 32], 32);
      c[0][j] = wmma_bf16(a0, b, c[0][j]);
      c[1][j] = wmma_bf16(a1, b, c[1][j]);
    }
    __syncthreads();  // all reads of Bs[buf] done before it is re-staged
    buf ^= 1;
  }

  const int l   = threadIdx.x & 31;
  const int col = l & 15;
  const int rb  = (l >> 4) << 3;

  if (mode == 0) {
    float* o = (float*)outp;
#pragma unroll
    for (int ms = 0; ms < 2; ++ms)
#pragma unroll
      for (int j = 0; j < 4; ++j)
#pragma unroll
        for (int r = 0; r < 8; ++r)
          o[(size_t)(m0 + ms * 16 + rb + r) * N + (n0 + j * 16 + col)] = c[ms][j][r];
  } else if (mode == 1) {
    __bf16* o = (__bf16*)outp;
#pragma unroll
    for (int ms = 0; ms < 2; ++ms)
#pragma unroll
      for (int j = 0; j < 4; ++j)
#pragma unroll
        for (int r = 0; r < 8; ++r) {
          int m = m0 + ms * 16 + rb + r, n = n0 + j * 16 + col;
          int b = m >> 11, t = m & (TSEQ - 1);
          int h = n >> 7, dd = n & (DATT - 1);
          o[(((size_t)(b * HEADS + h) * TSEQ) + t) * DATT + dd] = (__bf16)c[ms][j][r];
        }
  } else {  // mode 2: Vt [B][DATT][T]
    __bf16* o = (__bf16*)outp;
#pragma unroll
    for (int ms = 0; ms < 2; ++ms)
#pragma unroll
      for (int j = 0; j < 4; ++j)
#pragma unroll
        for (int r = 0; r < 8; ++r) {
          int m = m0 + ms * 16 + rb + r, n = n0 + j * 16 + col;
          int b = m >> 11, t = m & (TSEQ - 1);
          o[((size_t)(b * DATT + n) * TSEQ) + t] = (__bf16)c[ms][j][r];
        }
  }
}

// ---------------------------------------------------------------------------
// Flash attention pass: one wave per (b, h, 16-row Q tile).  Causal, online
// softmax over 32-key tiles.  Q prescaled by 1/sqrt(d) (folded into Wq).
// Emits per-head O [B,H,T,DATT] f32 plus row max M and row sum L [B,H,T].
// ---------------------------------------------------------------------------
__global__ void flash_attn_kernel(const __bf16* __restrict__ Q,
                                  const __bf16* __restrict__ Kc,
                                  const __bf16* __restrict__ Vt,
                                  float* __restrict__ O,
                                  float* __restrict__ Mrow,
                                  float* __restrict__ Lrow) {
  const int tilesPerHead = TSEQ / 16;
  const int bh = blockIdx.x / tilesPerHead;
  const int t0 = (blockIdx.x % tilesPerHead) * 16;
  const int b  = bh / HEADS;

  const __bf16* Qh  = Q  + (size_t)bh * TSEQ * DATT;
  const __bf16* Kh  = Kc + (size_t)bh * TSEQ * DATT;
  const __bf16* Vtb = Vt + (size_t)b * DATT * TSEQ;

  __shared__ alignas(16) __bf16 Pld[16 * 32];

  const int l   = threadIdx.x & 31;
  const int col = l & 15;
  const int rb  = (l >> 4) << 3;

  v16bf qf[4];
#pragma unroll
  for (int kc = 0; kc < 4; ++kc)
    qf[kc] = load_a_frag(Qh + (size_t)t0 * DATT + kc * 32, DATT);

  float m8[8], l8[8];
  v8f o[8];
#pragma unroll
  for (int r = 0; r < 8; ++r) { m8[r] = -3.0e38f; l8[r] = 0.0f; }
#pragma unroll
  for (int j = 0; j < 8; ++j) o[j] = v8f{};

  for (int s0 = 0; s0 <= t0 + 15; s0 += 32) {
    if (s0 + 32 <= t0 + 15) {  // hide next K tile's latency under this tile
      __builtin_prefetch(Kh + (size_t)(s0 + 32 + (l & 31)) * DATT, 0, 0);
    }
    // S = Q @ K^T for this 16x32 tile
    v8f s[2] = {v8f{}, v8f{}};
#pragma unroll
    for (int kc = 0; kc < 4; ++kc) {
#pragma unroll
      for (int ns = 0; ns < 2; ++ns) {
        v16bf bf = load_b_frag(Kh + (size_t)(s0 + ns * 16) * DATT + kc * 32, DATT);
        s[ns] = wmma_bf16(qf[kc], bf, s[ns]);
      }
    }
    // causal mask
#pragma unroll
    for (int ns = 0; ns < 2; ++ns)
#pragma unroll
      for (int r = 0; r < 8; ++r) {
        int t = t0 + rb + r, si = s0 + ns * 16 + col;
        if (si > t) s[ns][r] = -3.0e38f;
      }
    // row max over 32 cols (cols live across the 16 lanes of each half)
    float rm[8];
#pragma unroll
    for (int r = 0; r < 8; ++r) rm[r] = fmaxf(s[0][r], s[1][r]);
#pragma unroll
    for (int off = 1; off <= 8; off <<= 1)
#pragma unroll
      for (int r = 0; r < 8; ++r)
        rm[r] = fmaxf(rm[r], __shfl_xor(rm[r], off, 32));

    float alpha[8];
#pragma unroll
    for (int r = 0; r < 8; ++r) {
      float mn = fmaxf(m8[r], rm[r]);
      alpha[r] = __expf(m8[r] - mn);
      m8[r] = mn;
    }
    float rs[8];
#pragma unroll
    for (int r = 0; r < 8; ++r) rs[r] = 0.0f;
#pragma unroll
    for (int ns = 0; ns < 2; ++ns)
#pragma unroll
      for (int r = 0; r < 8; ++r) {
        float p = __expf(s[ns][r] - m8[r]);
        s[ns][r] = p;
        rs[r] += p;
      }
#pragma unroll
    for (int off = 1; off <= 8; off <<= 1)
#pragma unroll
      for (int r = 0; r < 8; ++r) rs[r] += __shfl_xor(rs[r], off, 32);
#pragma unroll
    for (int r = 0; r < 8; ++r) l8[r] = l8[r] * alpha[r] + rs[r];
#pragma unroll
    for (int j = 0; j < 8; ++j)
#pragma unroll
      for (int r = 0; r < 8; ++r) o[j][r] *= alpha[r];

    // Stage P through LDS to re-fragment C-layout -> A-layout (bf16).
    __syncthreads();
#pragma unroll
    for (int ns = 0; ns < 2; ++ns)
#pragma unroll
      for (int r = 0; r < 8; ++r)
        Pld[(rb + r) * 32 + ns * 16 + col] = (__bf16)s[ns][r];
    __syncthreads();
    v16bf pf = load_a_frag(Pld, 32);

    // O += P @ V  (V transposed in memory -> contiguous-in-key B fragments)
#pragma unroll
    for (int j = 0; j < 8; ++j) {
      v16bf bv = load_b_frag(Vtb + (size_t)(j * 16) * TSEQ + s0, TSEQ);
      o[j] = wmma_bf16(pf, bv, o[j]);
    }
  }

  float inv[8];
#pragma unroll
  for (int r = 0; r < 8; ++r) inv[r] = 1.0f / l8[r];
  float* Oh = O + (size_t)bh * TSEQ * DATT;
#pragma unroll
  for (int j = 0; j < 8; ++j)
#pragma unroll
    for (int r = 0; r < 8; ++r)
      Oh[(size_t)(t0 + rb + r) * DATT + j * 16 + col] = o[j][r] * inv[r];
  if (col == 0) {
#pragma unroll
    for (int r = 0; r < 8; ++r) {
      Mrow[(size_t)bh * TSEQ + t0 + rb + r] = m8[r];
      Lrow[(size_t)bh * TSEQ + t0 + rb + r] = l8[r];
    }
  }
}

// ---------------------------------------------------------------------------
// Attention-weights pass: recompute scores per head with WMMA, normalize with
// stored (m, l), average over heads -> attn [B,T,T].  One wave per 16x32 tile.
// ---------------------------------------------------------------------------
__global__ void attn_weights_kernel(const __bf16* __restrict__ Q,
                                    const __bf16* __restrict__ Kc,
                                    const float* __restrict__ Mrow,
                                    const float* __restrict__ Lrow,
                                    float* __restrict__ attnO) {
  const int s0 = blockIdx.x * 32;
  const int t0 = blockIdx.y * 16;
  const int b  = blockIdx.z;

  const int l   = threadIdx.x & 31;
  const int col = l & 15;
  const int rb  = (l >> 4) << 3;

  if (s0 > t0 + 15) {  // fully above diagonal: exact zeros
#pragma unroll
    for (int ns = 0; ns < 2; ++ns)
#pragma unroll
      for (int r = 0; r < 8; ++r)
        attnO[((size_t)b * TSEQ + (t0 + rb + r)) * TSEQ + (s0 + ns * 16 + col)] = 0.0f;
    return;
  }

  float acc[2][8];
#pragma unroll
  for (int ns = 0; ns < 2; ++ns)
#pragma unroll
    for (int r = 0; r < 8; ++r) acc[ns][r] = 0.0f;

  for (int h = 0; h < HEADS; ++h) {
    const int bh = b * HEADS + h;
    const __bf16* Qh = Q  + (size_t)bh * TSEQ * DATT;
    const __bf16* Kh = Kc + (size_t)bh * TSEQ * DATT;

    v8f s[2] = {v8f{}, v8f{}};
#pragma unroll
    for (int kc = 0; kc < 4; ++kc) {
      v16bf a = load_a_frag(Qh + (size_t)t0 * DATT + kc * 32, DATT);
#pragma unroll
      for (int ns = 0; ns < 2; ++ns) {
        v16bf bf = load_b_frag(Kh + (size_t)(s0 + ns * 16) * DATT + kc * 32, DATT);
        s[ns] = wmma_bf16(a, bf, s[ns]);
      }
    }
    float mv[8], li[8];
#pragma unroll
    for (int r = 0; r < 8; ++r) {
      mv[r] = Mrow[(size_t)bh * TSEQ + t0 + rb + r];
      li[r] = 1.0f / Lrow[(size_t)bh * TSEQ + t0 + rb + r];
    }
#pragma unroll
    for (int ns = 0; ns < 2; ++ns)
#pragma unroll
      for (int r = 0; r < 8; ++r) {
        int t = t0 + rb + r, si = s0 + ns * 16 + col;
        float p = (si <= t) ? __expf(s[ns][r] - mv[r]) * li[r] : 0.0f;
        acc[ns][r] += p * (1.0f / HEADS);
      }
  }
#pragma unroll
  for (int ns = 0; ns < 2; ++ns)
#pragma unroll
    for (int r = 0; r < 8; ++r)
      attnO[((size_t)b * TSEQ + (t0 + rb + r)) * TSEQ + (s0 + ns * 16 + col)] =
          acc[ns][r];
}

// mean over heads -> bf16 [B*T, DATT]
__global__ void mean_o_kernel(const float* __restrict__ O,
                              __bf16* __restrict__ meanO) {
  int i = blockIdx.x * blockDim.x + threadIdx.x;
  if (i >= BSZ * TSEQ * DATT) return;
  int dd = i & (DATT - 1);
  int bt = i >> 7;
  int t = bt & (TSEQ - 1);
  int b = bt >> 11;
  float s = 0.0f;
#pragma unroll
  for (int h = 0; h < HEADS; ++h)
    s += O[((size_t)(b * HEADS + h) * TSEQ + t) * DATT + dd];
  meanO[i] = (__bf16)(s * (1.0f / HEADS));
}

// ---------------------------------------------------------------------------
extern "C" void kernel_launch(void* const* d_in, const int* in_sizes, int n_in,
                              void* d_out, int out_size, void* d_ws, size_t ws_size,
                              hipStream_t stream) {
  (void)in_sizes; (void)n_in; (void)out_size; (void)ws_size;
  const float* x  = (const float*)d_in[0];
  const float* Wq = (const float*)d_in[1];
  const float* Wk = (const float*)d_in[2];
  const float* Wv = (const float*)d_in[3];
  const float* Wo = (const float*)d_in[4];

  char* ws = (char*)d_ws;
  size_t off = 0;
  auto alloc = [&](size_t bytes) -> char* {
    char* p = ws + off;
    off = (off + bytes + 255) & ~(size_t)255;
    return p;
  };
  const size_t NTOK = (size_t)BSZ * TSEQ;  // 8192
  __bf16* xb    = (__bf16*)alloc(NTOK * UNITS * 2);
  __bf16* WqT   = (__bf16*)alloc((size_t)UNITS * UNITS * 2);
  __bf16* WkT   = (__bf16*)alloc((size_t)UNITS * UNITS * 2);
  __bf16* WvT   = (__bf16*)alloc((size_t)DATT * UNITS * 2);
  __bf16* WoT   = (__bf16*)alloc((size_t)UNITS * DATT * 2);
  __bf16* Qb    = (__bf16*)alloc(NTOK * UNITS * 2);
  __bf16* Kb    = (__bf16*)alloc(NTOK * UNITS * 2);
  __bf16* Vt    = (__bf16*)alloc((size_t)BSZ * DATT * TSEQ * 2);
  __bf16* meanO = (__bf16*)alloc(NTOK * DATT * 2);
  float*  Obuf  = (float*)alloc(NTOK * UNITS * 4);  // [B,H,T,DATT]
  float*  Mrow  = (float*)alloc((size_t)BSZ * HEADS * TSEQ * 4);
  float*  Lrow  = (float*)alloc((size_t)BSZ * HEADS * TSEQ * 4);

  float* outMain = (float*)d_out;           // [B,T,UNITS]
  float* attnOut = outMain + NTOK * UNITS;  // [B,T,T]

  const float qscale = 0.088388347648318447f;  // 1/sqrt(128), folded into Wq

  // 1) casts / transposes
  {
    int n = (int)(NTOK * UNITS);
    cast_f32_bf16_kernel<<<(n + 255) / 256, 256, 0, stream>>>(x, xb, n);
  }
  transpose_cast_kernel<<<(UNITS * UNITS + 255) / 256, 256, 0, stream>>>(
      Wq, WqT, UNITS, UNITS, qscale);
  transpose_cast_kernel<<<(UNITS * UNITS + 255) / 256, 256, 0, stream>>>(
      Wk, WkT, UNITS, UNITS, 1.0f);
  transpose_cast_kernel<<<(UNITS * DATT + 255) / 256, 256, 0, stream>>>(
      Wv, WvT, UNITS, DATT, 1.0f);
  transpose_cast_kernel<<<(DATT * UNITS + 255) / 256, 256, 0, stream>>>(
      Wo, WoT, DATT, UNITS, 1.0f);

  // 2) projections (WMMA GEMMs, async-LDS staged B tiles)
  gemm_bf16_kernel<<<dim3(32, 16), 256, 0, stream>>>(xb, WqT, UNITS, UNITS, Qb, 1);
  gemm_bf16_kernel<<<dim3(32, 16), 256, 0, stream>>>(xb, WkT, UNITS, UNITS, Kb, 1);
  gemm_bf16_kernel<<<dim3(32, 2),  256, 0, stream>>>(xb, WvT, DATT,  UNITS, Vt, 2);

  // 3) flash attention (per-head O, row stats)
  flash_attn_kernel<<<BSZ * HEADS * (TSEQ / 16), 32, 0, stream>>>(
      Qb, Kb, Vt, Obuf, Mrow, Lrow);

  // 4) head-averaged attention weights
  attn_weights_kernel<<<dim3(TSEQ / 32, TSEQ / 16, BSZ), 32, 0, stream>>>(
      Qb, Kb, Mrow, Lrow, attnOut);

  // 5) mean over heads + output projection
  {
    int n = (int)(NTOK * DATT);
    mean_o_kernel<<<(n + 255) / 256, 256, 0, stream>>>(Obuf, meanO);
  }
  gemm_bf16_kernel<<<dim3(32, 16), 256, 0, stream>>>(meanO, WoT, UNITS, DATT,
                                                     outMain, 0);
}
